// PNAREG_18459769438674
// MI455X (gfx1250) — matrix operand order repaired and checked by reading, not compiled, not run on hardware
//
#include <hip/hip_runtime.h>
#include <hip/hip_bf16.h>

typedef __attribute__((ext_vector_type(16))) _Float16 v16h;
typedef __attribute__((ext_vector_type(8)))  _Float16 v8h;
typedef __attribute__((ext_vector_type(8)))  float    v8f;
typedef __attribute__((ext_vector_type(4)))  unsigned int v4u;
typedef __attribute__((ext_vector_type(8)))  unsigned int v8u;

#define F 64
#define LN_EPS 1e-5f
static constexpr float AVG_LOG_INV = 1.0f / 2.8332133440562162f; // 1/log(17)

// ---- f16 fragment loads (16B-aligned pieces; CDNA5 wave32 A/B layouts) ----
__device__ __forceinline__ v16h ld2x8(const _Float16* p0, const _Float16* p1) {
  v8h lo = *(const v8h*)p0;
  v8h hi = *(const v8h*)p1;
  return __builtin_shufflevector(lo, hi, 0,1,2,3,4,5,6,7,8,9,10,11,12,13,14,15);
}
__device__ __forceinline__ v16h ld16c(const _Float16* p) { return ld2x8(p, p + 8); }

// ---- Tensor Data Mover: stage BT [64][Ka] f16 (row-major) into LDS ----
// D# per CDNA5 ISA: group0 = {flags, lds_addr, global_addr, type=2},
// group1 = {data_size=2B, tensor_dim0=Ka, tensor_dim1=64, tile_dim0=Ka, tile_dim1=64,
//           tensor_dim0_stride=Ka}; groups 2/3 zero (<=2D tile).
__device__ __forceinline__ void tdm_stage_bt(const void* gsrc, unsigned lds_off, unsigned ka) {
  unsigned long long ga = (unsigned long long)(uintptr_t)gsrc;
  v4u g0;
  g0[0] = 1u;                                              // count=1 (valid user D#)
  g0[1] = lds_off;                                         // lds_addr (bytes)
  g0[2] = (unsigned)ga;                                    // global_addr[31:0]
  g0[3] = (unsigned)((ga >> 32) & 0x01FFFFFFu) | (2u << 30); // addr[56:32] | type=2
  v8u g1;
  g1[0] = 1u << 16;                                        // wg_mask=0, data_size=1 (2 bytes)
  g1[1] = (ka & 0xFFFFu) << 16;                            // tensor_dim0[15:0]
  g1[2] = ((ka >> 16) & 0xFFFFu) | (64u << 16);            // tensor_dim0[31:16] | tensor_dim1[15:0]
  g1[3] = (ka & 0xFFFFu) << 16;                            // tensor_dim1[31:16]=0 | tile_dim0=Ka
  g1[4] = 64u;                                             // tile_dim1=64, tile_dim2=0
  g1[5] = ka;                                              // tensor_dim0_stride[31:0]
  g1[6] = 0u;                                              // stride0 hi | dim1_stride lo
  g1[7] = 0u;                                              // dim1_stride hi
  v4u g2 = {0u, 0u, 0u, 0u};
  v4u g3 = {0u, 0u, 0u, 0u};
  asm volatile("tensor_load_to_lds %0, %1, %2, %3"
               :: "s"(g0), "s"(g1), "s"(g2), "s"(g3)
               : "memory");
  __builtin_amdgcn_s_wait_tensorcnt(0);
}

// ---- float atomic max/min via ordered int tricks (no NaNs in this pipeline) ----
__device__ __forceinline__ void atomicMaxF(float* a, float v) {
  if (!(__float_as_uint(v) & 0x80000000u)) atomicMax((int*)a, __float_as_int(v));
  else                                     atomicMin((unsigned int*)a, __float_as_uint(v));
}
__device__ __forceinline__ void atomicMinF(float* a, float v) {
  if (!(__float_as_uint(v) & 0x80000000u)) atomicMin((int*)a, __float_as_int(v));
  else                                     atomicMax((unsigned int*)a, __float_as_uint(v));
}

// ---------------- small utility kernels ----------------
__global__ void k_zero_f32(float* __restrict__ p, int n) {
  int t = blockIdx.x * blockDim.x + threadIdx.x;
  if (t < n) p[t] = 0.f;
}
__global__ void k_scatter_count(const int* __restrict__ idx, float* __restrict__ out, int n) {
  int t = blockIdx.x * blockDim.x + threadIdx.x;
  if (t < n) atomicAdd(&out[idx[t]], 1.f);
}
__global__ void k_f32_to_f16(const float* __restrict__ s, _Float16* __restrict__ d, int n) {
  int t = blockIdx.x * blockDim.x + threadIdx.x;
  if (t < n) d[t] = (_Float16)s[t];
}
// w: [K][64] f32 row-major -> wT: [64][K] f16 (contiguous K per output column)
__global__ void k_transposeT(const float* __restrict__ w, _Float16* __restrict__ wT, int K) {
  int t = blockIdx.x * blockDim.x + threadIdx.x;
  if (t >= K * 64) return;
  int k = t >> 6, n = t & 63;
  wT[(size_t)n * K + k] = (_Float16)w[t];
}
__global__ void k_init_agg(float* __restrict__ s, float* __restrict__ sq,
                           float* __restrict__ mx, float* __restrict__ mn, int n) {
  int t = blockIdx.x * blockDim.x + threadIdx.x;
  if (t >= n) return;
  s[t] = 0.f; sq[t] = 0.f;
  mx[t] = -__builtin_huge_valf();
  mn[t] =  __builtin_huge_valf();
}

// ---------------- edge message GEMM + scatter aggregation ----------------
// One wave per 16 edges: m[16,64] = [h[dst]|h[src]](f16)[16,128] @ preT^T + pre_b
// BT (preT, 16KB) staged into LDS once per block via TDM.
__global__ void k_edge_msg(const _Float16* __restrict__ h_h,
                           const int* __restrict__ esrc, const int* __restrict__ edst,
                           const _Float16* __restrict__ preT,   // [64][128] f16
                           const float* __restrict__ pre_b,
                           float* __restrict__ sum, float* __restrict__ sumsq,
                           float* __restrict__ mx, float* __restrict__ mn,
                           int ntiles) {
  extern __shared__ _Float16 Bsh[] __attribute__((aligned(16)));
  if (threadIdx.x < 32)   // wave 0 issues the TDM descriptor and waits TENSORcnt
    tdm_stage_bt(preT, (unsigned)__builtin_amdgcn_groupstaticsize(), 128u);
  __syncthreads();

  int tile = blockIdx.x * (blockDim.x >> 5) + (threadIdx.x >> 5);
  if (tile >= ntiles) return;
  int lane = threadIdx.x & 31;
  int l15  = lane & 15;
  bool hig = lane >= 16;
  int base = hig ? 16 : 0;

  int e = tile * 16 + l15;
  const _Float16* hd = h_h + (size_t)edst[e] * F;  // x_i = dst
  const _Float16* hs = h_h + (size_t)esrc[e] * F;  // x_j = src

  v8f acc[4] = {};
  #pragma unroll
  for (int kc = 0; kc < 4; ++kc) {             // K = 128 in 4 chunks of 32
    const _Float16* p = (kc < 2) ? hd : hs;
    int c = (kc & 1) * 32;
    v16h a = ld2x8(p + c + base, p + c + base + 8);
    int kb = kc * 32 + (hig ? 16 : 0);
    #pragma unroll
    for (int nt = 0; nt < 4; ++nt) {
      v16h b = ld16c(&Bsh[(size_t)(nt * 16 + l15) * 128 + kb]);
      acc[nt] = __builtin_amdgcn_wmma_f32_16x16x32_f16(
          false, a, false, b, (short)0, acc[nt], false, false);
    }
  }

  int d2[8];
  #pragma unroll
  for (int v = 0; v < 8; ++v) d2[v] = edst[tile * 16 + v + (hig ? 8 : 0)];

  #pragma unroll
  for (int nt = 0; nt < 4; ++nt) {
    int n = nt * 16 + l15;
    float bb = pre_b[n];
    #pragma unroll
    for (int v = 0; v < 8; ++v) {
      float m = acc[nt][v] + bb;
      size_t o = (size_t)d2[v] * F + n;
      atomicAdd(sum + o, m);
      atomicAdd(sumsq + o, m * m);
      atomicMaxF(mx + o, m);
      atomicMinF(mn + o, m);
    }
  }
}

// ---------------- finalize aggregation: build 13F concat row in f16 ----------------
__global__ void k_build_a13(const float* __restrict__ h, const float* __restrict__ sum,
                            const float* __restrict__ sumsq, const float* __restrict__ mx,
                            const float* __restrict__ mn, const float* __restrict__ degF,
                            _Float16* __restrict__ a13, int total) {
  int t = blockIdx.x * blockDim.x + threadIdx.x;
  if (t >= total) return;
  int n = t >> 6, c = t & 63;
  float cnt  = degF[n];
  float deg  = fmaxf(cnt, 1.f);
  float inv  = 1.f / deg;
  float mean = sum[t] * inv;
  float msq  = sumsq[t] * inv;
  float sd   = sqrtf(fmaxf(msq - mean * mean, 0.f) + 1e-5f);
  float mxv  = (cnt > 0.f) ? mx[t] : 0.f;   // empty segment -> 0 (ref isfinite fixup)
  float mnv  = (cnt > 0.f) ? mn[t] : 0.f;
  float amp  = logf(deg + 1.f) * AVG_LOG_INV;
  float ia   = 1.f / amp;
  _Float16* r = a13 + (size_t)n * 832 + c;
  r[0]   = (_Float16)h[t];
  r[64]  = (_Float16)mean;        r[128] = (_Float16)mxv;
  r[192] = (_Float16)mnv;         r[256] = (_Float16)sd;
  r[320] = (_Float16)(mean*amp);  r[384] = (_Float16)(mxv*amp);
  r[448] = (_Float16)(mnv*amp);   r[512] = (_Float16)(sd*amp);
  r[576] = (_Float16)(mean*ia);   r[640] = (_Float16)(mxv*ia);
  r[704] = (_Float16)(mnv*ia);    r[768] = (_Float16)(sd*ia);
}

// ---------------- generic 16-row x 64-col WMMA GEMM, fused epilogue ----------------
// B staged in LDS via TDM once per block; A fragments streamed from global.
template <bool RELU, bool LN, bool OUTF16>
__global__ void k_gemm(const _Float16* __restrict__ A,   // [M][Ka] f16
                       const _Float16* __restrict__ BT,  // [64][Ka] f16
                       const float* __restrict__ bias,
                       const float* __restrict__ lng, const float* __restrict__ lnb,
                       float* __restrict__ outF, _Float16* __restrict__ outH,
                       int Ka, int ntiles) {
  extern __shared__ _Float16 Bsh[] __attribute__((aligned(16)));
  if (threadIdx.x < 32)
    tdm_stage_bt(BT, (unsigned)__builtin_amdgcn_groupstaticsize(), (unsigned)Ka);
  __syncthreads();

  int tile = blockIdx.x * (blockDim.x >> 5) + (threadIdx.x >> 5);
  if (tile >= ntiles) return;
  int lane = threadIdx.x & 31;
  int l15  = lane & 15;
  bool hig = lane >= 16;
  int base = hig ? 16 : 0;

  const _Float16* arow = A + (size_t)(tile * 16 + l15) * Ka;
  int nk = Ka >> 5;
  v8f acc[4] = {};
  for (int kc = 0; kc < nk; ++kc) {
    int c = kc * 32;
    v16h a = ld2x8(arow + c + base, arow + c + base + 8);
    int kb = c + (hig ? 16 : 0);
    #pragma unroll
    for (int nt = 0; nt < 4; ++nt) {
      v16h b = ld16c(&Bsh[(size_t)(nt * 16 + l15) * Ka + kb]);
      acc[nt] = __builtin_amdgcn_wmma_f32_16x16x32_f16(
          false, a, false, b, (short)0, acc[nt], false, false);
    }
  }

  float y[4][8];
  #pragma unroll
  for (int nt = 0; nt < 4; ++nt) {
    float bb = bias[nt * 16 + l15];
    #pragma unroll
    for (int v = 0; v < 8; ++v) {
      float tv = acc[nt][v] + bb;
      if (RELU) tv = fmaxf(tv, 0.f);
      y[nt][v] = tv;
    }
  }
  if (LN) {
    float g[4], b2[4];
    #pragma unroll
    for (int nt = 0; nt < 4; ++nt) { g[nt] = lng[nt*16+l15]; b2[nt] = lnb[nt*16+l15]; }
    #pragma unroll
    for (int v = 0; v < 8; ++v) {
      float s1 = 0.f, s2 = 0.f;
      #pragma unroll
      for (int nt = 0; nt < 4; ++nt) { s1 += y[nt][v]; s2 += y[nt][v]*y[nt][v]; }
      #pragma unroll
      for (int m = 1; m < 16; m <<= 1) {  // reduce within each 16-lane half (one row group)
        s1 += __shfl_xor(s1, m, 32);
        s2 += __shfl_xor(s2, m, 32);
      }
      float mu  = s1 * (1.f / 64.f);
      float var = s2 * (1.f / 64.f) - mu * mu;
      float rs  = rsqrtf(var + LN_EPS);
      #pragma unroll
      for (int nt = 0; nt < 4; ++nt) y[nt][v] = (y[nt][v] - mu) * rs * g[nt] + b2[nt];
    }
  }
  #pragma unroll
  for (int v = 0; v < 8; ++v) {
    size_t row = (size_t)(tile * 16 + v + (hig ? 8 : 0)) * F;
    #pragma unroll
    for (int nt = 0; nt < 4; ++nt) {
      int n = nt * 16 + l15;
      if (OUTF16) outH[row + n] = (_Float16)y[nt][v];
      else        outF[row + n] = y[nt][v];
    }
  }
}

// ---------------- pooling + MLP head ----------------
__global__ void k_pool_accum(const float* __restrict__ h, const int* __restrict__ batch,
                             float* __restrict__ pool, int total) {
  int t = blockIdx.x * blockDim.x + threadIdx.x;
  if (t >= total) return;
  int n = t >> 6, c = t & 63;
  atomicAdd(&pool[(size_t)batch[n] * F + c], h[t]);
}
__global__ void k_mlp_head(const float* __restrict__ pool, const float* __restrict__ gcnt,
                           const float* __restrict__ w1, const float* __restrict__ b1,
                           const float* __restrict__ w2, const float* __restrict__ b2,
                           float* __restrict__ out, int G) {
  int g = blockIdx.x * blockDim.x + threadIdx.x;
  if (g >= G) return;
  float inv = 1.f / fmaxf(gcnt[g], 1.f);
  const float* p = pool + (size_t)g * F;
  float o = b2[0];
  for (int j = 0; j < 32; ++j) {
    float acc = b1[j];
    for (int k = 0; k < F; ++k) acc += p[k] * inv * w1[k * 32 + j];
    o += fmaxf(acc, 0.f) * w2[j];
  }
  out[g] = o;
}

extern "C" void kernel_launch(void* const* d_in, const int* in_sizes, int n_in,
                              void* d_out, int out_size, void* d_ws, size_t ws_size,
                              hipStream_t stream) {
  (void)n_in; (void)ws_size;
  const float* x      = (const float*)d_in[0];
  const int*   esrc   = (const int*)d_in[1];
  const int*   edst   = (const int*)d_in[2];
  const int*   batch  = (const int*)d_in[3];
  const float* pre_w  = (const float*)d_in[4];
  const float* pre_b  = (const float*)d_in[5];
  const float* post_w = (const float*)d_in[6];
  const float* post_b = (const float*)d_in[7];
  const float* lin_w  = (const float*)d_in[8];
  const float* lin_b  = (const float*)d_in[9];
  const float* ln_w   = (const float*)d_in[10];
  const float* ln_b   = (const float*)d_in[11];
  const float* mlp1w  = (const float*)d_in[12];
  const float* mlp1b  = (const float*)d_in[13];
  const float* mlp2w  = (const float*)d_in[14];
  const float* mlp2b  = (const float*)d_in[15];

  const int E = in_sizes[1];
  const int N = in_sizes[3];
  const int G = out_size;                  // 500
  const size_t N64 = (size_t)N * F;

  // carve workspace (all L2-resident: ~173 MB total)
  char* wsp = (char*)d_ws;
  auto take = [&](size_t bytes) { char* p = wsp; wsp += (bytes + 255) & ~(size_t)255; return p; };
  float*    hA    = (float*)take(N64 * 4);
  float*    hB    = (float*)take(N64 * 4);
  _Float16* h_h   = (_Float16*)take(N64 * 2);
  _Float16* t1    = (_Float16*)take(N64 * 2);
  float*    sumB  = (float*)take(N64 * 4);
  float*    sqB   = (float*)take(N64 * 4);
  float*    mxB   = (float*)take(N64 * 4);
  float*    mnB   = (float*)take(N64 * 4);
  _Float16* a13   = (_Float16*)take((size_t)N * 832 * 2);
  float*    degF  = (float*)take((size_t)N * 4);
  _Float16* preT  = (_Float16*)take((size_t)64 * 128 * 2);
  _Float16* postT = (_Float16*)take((size_t)64 * 832 * 2);
  _Float16* linT  = (_Float16*)take((size_t)64 * 64 * 2);
  float*    pool  = (float*)take((size_t)G * F * 4);
  float*    gcnt  = (float*)take((size_t)G * 4);

  const int TB = 256;
  auto nb = [](long long n, int b) { return (unsigned)((n + b - 1) / b); };

  // degrees (same for all layers)
  k_zero_f32<<<nb(N, TB), TB, 0, stream>>>(degF, N);
  k_scatter_count<<<nb(E, TB), TB, 0, stream>>>(edst, degF, E);

  const int etiles = E / 16, ntiles = N / 16;
  const float* hin = x;
  float* hout = hA;

  for (int i = 0; i < 3; ++i) {
    k_transposeT<<<nb(128 * 64, TB), TB, 0, stream>>>(pre_w  + (size_t)i * 128 * 64, preT, 128);
    k_transposeT<<<nb(832 * 64, TB), TB, 0, stream>>>(post_w + (size_t)i * 832 * 64, postT, 832);
    k_transposeT<<<nb(64 * 64,  TB), TB, 0, stream>>>(lin_w  + (size_t)i * 64 * 64,  linT, 64);
    k_f32_to_f16<<<nb((long long)N64, TB), TB, 0, stream>>>(hin, h_h, (int)N64);
    k_init_agg<<<nb((long long)N64, TB), TB, 0, stream>>>(sumB, sqB, mxB, mnB, (int)N64);

    k_edge_msg<<<nb(etiles, 4), 128, (size_t)64 * 128 * 2, stream>>>(
        h_h, esrc, edst, preT, pre_b + (size_t)i * F, sumB, sqB, mxB, mnB, etiles);

    k_build_a13<<<nb((long long)N64, TB), TB, 0, stream>>>(hin, sumB, sqB, mxB, mnB, degF,
                                                           a13, (int)N64);

    // post_nn: [N,832] x [832,64] -> t1 (f16); BT staged via TDM (106KB LDS)
    k_gemm<false, false, true><<<nb(ntiles, 4), 128, (size_t)64 * 832 * 2, stream>>>(
        a13, postT, post_b + (size_t)i * F, nullptr, nullptr, nullptr, t1, 832, ntiles);

    // lin: [N,64] x [64,64] -> h_out (f32), fused ReLU (+LN except last layer)
    if (i < 2) {
      k_gemm<true, true, false><<<nb(ntiles, 4), 128, (size_t)64 * 64 * 2, stream>>>(
          t1, linT, lin_b + (size_t)i * F, ln_w + (size_t)i * F, ln_b + (size_t)i * F,
          hout, nullptr, 64, ntiles);
    } else {
      k_gemm<true, false, false><<<nb(ntiles, 4), 128, (size_t)64 * 64 * 2, stream>>>(
          t1, linT, lin_b + (size_t)i * F, nullptr, nullptr,
          hout, nullptr, 64, ntiles);
    }
    hin  = hout;
    hout = (hout == hA) ? hB : hA;
  }

  // global mean pool + MLP head
  k_zero_f32<<<nb((long long)G * F, TB), TB, 0, stream>>>(pool, G * F);
  k_zero_f32<<<nb(G, TB), TB, 0, stream>>>(gcnt, G);
  k_scatter_count<<<nb(N, TB), TB, 0, stream>>>(batch, gcnt, N);
  k_pool_accum<<<nb((long long)N64, TB), TB, 0, stream>>>(hin, batch, pool, (int)N64);
  k_mlp_head<<<nb(G, 128), 128, 0, stream>>>(pool, gcnt, mlp1w, mlp1b, mlp2w, mlp2b,
                                             (float*)d_out, G);
}
